// MyLLMFFNCoE_55250459295818
// MI455X (gfx1250) — matrix operand order repaired and checked by reading, not compiled, not run on hardware
//
#include <hip/hip_runtime.h>

// ---------------- types ----------------
typedef __bf16 bf16_t;
typedef bf16_t v16bf __attribute__((ext_vector_type(16)));
typedef bf16_t v8bf  __attribute__((ext_vector_type(8)));
typedef float  v8f   __attribute__((ext_vector_type(8)));

// ---------------- problem constants ----------------
#define T_TOK 4096   // B*L tokens
#define HD    2048   // hidden
#define ED    1024   // expert size
#define NEXP  6      // routed experts
#define SED   2048   // S*E shared intermediate

// ================= utility kernels =================

__global__ void __launch_bounds__(256) k_f2bf(const float* __restrict__ s,
                                              bf16_t* __restrict__ d, int n) {
  int i = blockIdx.x * 256 + threadIdx.x;
  if (i < n) d[i] = (bf16_t)s[i];
}

// tiled transpose-convert: src f32 [K x N] row-major -> dst bf16 [N x K], batched over z.
// 32x32 tiles via LDS: coalesced reads and writes. K, N multiples of 32.
__global__ void __launch_bounds__(256) k_tconv(const float* __restrict__ s,
                                               bf16_t* __restrict__ d, int K, int N) {
  __shared__ float tile[32][33];
  long total = (long)K * N;
  s += (long)blockIdx.z * total;
  d += (long)blockIdx.z * total;
  int tx = threadIdx.x & 31, ty = threadIdx.x >> 5;  // 32 x 8
  int k0 = blockIdx.x * 32, n0 = blockIdx.y * 32;
#pragma unroll
  for (int i = 0; i < 4; ++i)
    tile[ty + i * 8][tx] = s[(long)(k0 + ty + i * 8) * N + (n0 + tx)];
  __syncthreads();
#pragma unroll
  for (int i = 0; i < 4; ++i)
    d[(long)(n0 + ty + i * 8) * K + (k0 + tx)] = (bf16_t)tile[tx][ty + i * 8];
}

// router: one wave per token. gate = x@Wg + bg over 8 experts, drop 2 smallest,
// softmax over remaining 6, write w[t][0..7] (cols 6,7 may be nonzero but unused).
__global__ void __launch_bounds__(256) k_router(const float* __restrict__ xt,
                                                const float* __restrict__ Wg,
                                                const float* __restrict__ bg,
                                                float* __restrict__ w) {
  int lane = threadIdx.x & 31;
  int t = blockIdx.x * 8 + (threadIdx.x >> 5);
  const float* xr = xt + (long)t * HD;
  float acc[8];
#pragma unroll
  for (int e = 0; e < 8; e++) acc[e] = 0.f;
  for (int h = lane; h < HD; h += 32) {
    float xv = xr[h];
    const float* wr = Wg + (long)h * 8;
#pragma unroll
    for (int e = 0; e < 8; e++) acc[e] += xv * wr[e];
  }
#pragma unroll
  for (int off = 16; off > 0; off >>= 1)
#pragma unroll
    for (int e = 0; e < 8; e++) acc[e] += __shfl_xor(acc[e], off, 32);
  if (lane == 0) {
    float g[8];
#pragma unroll
    for (int e = 0; e < 8; e++) g[e] = acc[e] + bg[e];
    int i1 = 0;
#pragma unroll
    for (int e = 1; e < 8; e++) if (g[e] < g[i1]) i1 = e;
    int i2 = -1;
#pragma unroll
    for (int e = 0; e < 8; e++) if (e != i1 && (i2 < 0 || g[e] < g[i2])) i2 = e;
    float m = -3.4e38f;
#pragma unroll
    for (int e = 0; e < 8; e++) if (e != i1 && e != i2) m = fmaxf(m, g[e]);
    float s = 0.f, ww[8];
#pragma unroll
    for (int e = 0; e < 8; e++) { ww[e] = (e == i1 || e == i2) ? 0.f : __expf(g[e] - m); s += ww[e]; }
    float inv = 1.f / s;
#pragma unroll
    for (int e = 0; e < 8; e++) w[(long)t * 8 + e] = ww[e] * inv;
  }
}

// residual: xf = res + out ; xb = bf16(xf)
__global__ void __launch_bounds__(256) k_residual(const float* __restrict__ res,
                                                  const float* __restrict__ out,
                                                  float* __restrict__ xf,
                                                  bf16_t* __restrict__ xb, int n) {
  int i = blockIdx.x * 256 + threadIdx.x;
  if (i < n) { float v = res[i] + out[i]; xf[i] = v; xb[i] = (bf16_t)v; }
}

// ================= WMMA GEMM =================
// C[M x N] = A[M x K](bf16, row-major, lda) * Bt[N x K](bf16 = B^T, ldbt) (+ epilogue)
// Block: 256 threads = 8 wave32s; block tile 128x128, K-step 64 (2 WMMA passes).
// Wave wid: rows (wid&3)*32 (two 16-row WMMA tiles), cols (wid>>2)*64 (four 16-col tiles)
// -> 16 WMMAs per wave per K-step (per wait/barrier round).
// Global->LDS feed: global_load_async_to_lds_b128, double-buffered, ASYNCcnt-tracked.

enum { EP_STORE = 0, EP_SWIGLU = 1, EP_ADDOUT = 2 };

#define BK 64
#define LS 72                 // LDS row stride in bf16 elems (64 + 8 pad -> 144B rows)
#define BUF_ELEMS (128 * LS)  // one tile buffer (18432 B)

__device__ __forceinline__ v16bf cat16(v8bf lo, v8bf hi) {
  return __builtin_shufflevector(lo, hi, 0, 1, 2, 3, 4, 5, 6, 7,
                                 8, 9, 10, 11, 12, 13, 14, 15);
}

// per-lane async 16B copy: global -> LDS (ISA: LDS addr = generic addr[31:0])
__device__ __forceinline__ void async_ld_b128(unsigned lds_addr, const bf16_t* gptr) {
  asm volatile("global_load_async_to_lds_b128 %0, %1, off"
               :: "v"(lds_addr), "v"(gptr) : "memory");
}
__device__ __forceinline__ void wait_async0() {
  asm volatile("s_wait_asynccnt 0" ::: "memory");
}
__device__ __forceinline__ unsigned lds_addr32(const void* p) {
  return (unsigned)(unsigned long long)p;
}

template <int EP>
__global__ void __launch_bounds__(256) k_gemm(
    const bf16_t* __restrict__ A, int lda,
    const bf16_t* __restrict__ Bt, int ldbt,
    const float* __restrict__ bias,
    void* __restrict__ Cv, int ldc,
    const bf16_t* __restrict__ hup,   // EP_SWIGLU: elementwise multiplier (same layout as C)
    const float* __restrict__ wvec,   // per-token scale, stride 8 (router weight col); may be null
    int Ktot) {
  __shared__ __align__(16) bf16_t Al[2 * BUF_ELEMS];
  __shared__ __align__(16) bf16_t Bl[2 * BUF_ELEMS];

  const int tid = threadIdx.x;
  const int lane = tid & 31;
  const int wid = tid >> 5;
  const int Mo = blockIdx.y * 128, No = blockIdx.x * 128;
  const int mbase = (wid & 3) * 32;   // wave's 32 rows within 128
  const int nbase = (wid >> 2) * 64;  // wave's 64 cols within 128

  v8f c[2][4];
#pragma unroll
  for (int i = 0; i < 2; ++i)
#pragma unroll
    for (int j = 0; j < 4; ++j) c[i][j] = (v8f){};

  // staging: 128 rows x 64 K per matrix = 1024 x 16B chunks; 4 chunks/thread/matrix
  const bf16_t* gA[4];
  const bf16_t* gB[4];
  unsigned lA[4], lB[4];
#pragma unroll
  for (int cc = 0; cc < 4; ++cc) {
    int idc = tid * 4 + cc;
    int rr = idc >> 3, kk = (idc & 7) * 8;
    gA[cc] = A + (long)(Mo + rr) * lda + kk;
    gB[cc] = Bt + (long)(No + rr) * ldbt + kk;
    lA[cc] = lds_addr32(&Al[rr * LS + kk]);
    lB[cc] = lds_addr32(&Bl[rr * LS + kk]);
  }

  // fragment addressing (ISA 16-bit A 16x32 / B 32x16 layouts)
  const int fm = lane & 15;
  const int kbA = (lane >> 4) * 8;
  const int kbB = (lane >> 4) * 8;

  // prologue: fill buffer 0
#pragma unroll
  for (int cc = 0; cc < 4; ++cc) {
    async_ld_b128(lA[cc], gA[cc]);
    async_ld_b128(lB[cc], gB[cc]);
  }

  int cur = 0;
  for (int k0 = 0; k0 < Ktot; k0 += BK) {
    wait_async0();      // this thread's chunks for buf[cur] are in LDS
    __syncthreads();    // everyone's chunks are in LDS

    if (k0 + BK < Ktot) {  // issue next tile into the other buffer
      const unsigned bo = (unsigned)((cur ^ 1) * BUF_ELEMS * 2);  // bytes
#pragma unroll
      for (int cc = 0; cc < 4; ++cc) {
        async_ld_b128(lA[cc] + bo, gA[cc] + k0 + BK);
        async_ld_b128(lB[cc] + bo, gB[cc] + k0 + BK);
      }
      __builtin_prefetch(gA[0] + k0 + 2 * BK, 0, 1);  // warm L2 one tile further
      __builtin_prefetch(gB[0] + k0 + 2 * BK, 0, 1);
    }

    const bf16_t* Ab = Al + cur * BUF_ELEMS;
    const bf16_t* Bb = Bl + cur * BUF_ELEMS;

#pragma unroll
    for (int s = 0; s < 2; ++s) {  // two 32-K WMMA passes over the staged 64-K tile
      const bf16_t* a0p = Ab + (mbase + fm) * LS + s * 32;
      const bf16_t* a1p = Ab + (mbase + 16 + fm) * LS + s * 32;
      v16bf a0 = cat16(*(const v8bf*)(a0p + kbA), *(const v8bf*)(a0p + 16 + kbA));
      v16bf a1 = cat16(*(const v8bf*)(a1p + kbA), *(const v8bf*)(a1p + 16 + kbA));

      v16bf b[4];
#pragma unroll
      for (int j = 0; j < 4; ++j) {
        const bf16_t* bp = Bb + (nbase + j * 16 + fm) * LS + s * 32 + kbB;
        b[j] = cat16(*(const v8bf*)bp, *(const v8bf*)(bp + 16));
      }

#pragma unroll
      for (int j = 0; j < 4; ++j) {
        c[0][j] = __builtin_amdgcn_wmma_f32_16x16x32_bf16(false, a0, false, b[j], (short)0, c[0][j], false, false);
        c[1][j] = __builtin_amdgcn_wmma_f32_16x16x32_bf16(false, a1, false, b[j], (short)0, c[1][j], false, false);
      }
    }

    __syncthreads();  // all waves done reading buf[cur] before it is refilled
    cur ^= 1;
  }

  // epilogue: C/D f32 layout: lane L -> col = L&15, VGPR r -> row = (L>>4)*8 + r
  const int en = lane & 15;
  const int em0 = (lane >> 4) * 8;
#pragma unroll
  for (int i = 0; i < 2; ++i) {
#pragma unroll
    for (int r8 = 0; r8 < 8; ++r8) {
      int gm = Mo + mbase + i * 16 + em0 + r8;
#pragma unroll
      for (int j = 0; j < 4; ++j) {
        int gn = No + nbase + j * 16 + en;
        float acc = c[i][j][r8];
        long idx = (long)gm * ldc + gn;
        float bb = bias[gn];
        if (EP == EP_STORE) {
          ((bf16_t*)Cv)[idx] = (bf16_t)(acc + bb);
        } else if (EP == EP_SWIGLU) {
          float v = acc + bb;
          float o = v * (1.f / (1.f + __expf(-v))) * (float)hup[idx];
          if (wvec) o *= wvec[(long)gm * 8];  // fold router weight into h
          ((bf16_t*)Cv)[idx] = (bf16_t)o;
        } else {  // EP_ADDOUT
          float add = acc + (wvec ? wvec[(long)gm * 8] * bb : bb);
          ((float*)Cv)[idx] += add;
        }
      }
    }
  }
}

// ================= launch =================
extern "C" void kernel_launch(void* const* d_in, const int* in_sizes, int n_in,
                              void* d_out, int out_size, void* d_ws, size_t ws_size,
                              hipStream_t stream) {
  const float* x   = (const float*)d_in[0];
  const float* Wg  = (const float*)d_in[1];
  const float* bg  = (const float*)d_in[2];
  const float* Wup = (const float*)d_in[3];
  const float* bup = (const float*)d_in[4];
  const float* Wsw = (const float*)d_in[5];
  const float* bsw = (const float*)d_in[6];
  const float* Wdn = (const float*)d_in[7];
  const float* bdn = (const float*)d_in[8];
  const float* Wsu = (const float*)d_in[9];
  const float* bsu = (const float*)d_in[10];
  const float* Wsa = (const float*)d_in[11];
  const float* bsa = (const float*)d_in[12];
  const float* Wsd = (const float*)d_in[13];
  const float* bsd = (const float*)d_in[14];

  char* p = (char*)d_ws;
  auto alloc = [&](size_t bytes) {
    char* r = p;
    p += (bytes + 255) & ~(size_t)255;
    return (void*)r;
  };

  // bf16 transposed weights (B^T layout: [Ncols][K])
  bf16_t* WupT = (bf16_t*)alloc((size_t)NEXP * ED * HD * 2);   // per expert [E][H]
  bf16_t* WswT = (bf16_t*)alloc((size_t)NEXP * ED * ED * 2);   // per expert [E][E]
  bf16_t* WdnT = (bf16_t*)alloc((size_t)NEXP * HD * ED * 2);   // per expert [H][E]
  bf16_t* WsuT = (bf16_t*)alloc((size_t)SED * HD * 2);         // [2048][H]
  bf16_t* WsaT = (bf16_t*)alloc((size_t)ED * ED * 2);          // [E][E]
  bf16_t* WsdT = (bf16_t*)alloc((size_t)HD * SED * 2);         // [H][2048]
  // activations
  bf16_t* xt_bf = (bf16_t*)alloc((size_t)T_TOK * HD * 2);
  bf16_t* hupB  = (bf16_t*)alloc((size_t)T_TOK * ED * 2);
  bf16_t* hB    = (bf16_t*)alloc((size_t)T_TOK * ED * 2);
  bf16_t* shB   = (bf16_t*)alloc((size_t)T_TOK * SED * 2);
  bf16_t* h2B   = (bf16_t*)alloc((size_t)T_TOK * SED * 2);
  float*  xtF   = (float*)alloc((size_t)T_TOK * HD * 4);
  float*  outF  = (float*)alloc((size_t)T_TOK * HD * 4);
  float*  wR    = (float*)alloc((size_t)T_TOK * 8 * 4);
  (void)ws_size; (void)n_in; (void)in_sizes; (void)out_size;

  // --- one-time weight transpose/convert (tiled, coalesced) ---
  k_tconv<<<dim3(HD / 32, ED / 32, NEXP), 256, 0, stream>>>(Wup, WupT, HD, ED);
  k_tconv<<<dim3(ED / 32, ED / 32, NEXP), 256, 0, stream>>>(Wsw, WswT, ED, ED);
  k_tconv<<<dim3(ED / 32, HD / 32, NEXP), 256, 0, stream>>>(Wdn, WdnT, ED, HD);
  k_tconv<<<dim3(HD / 32, SED / 32, 1), 256, 0, stream>>>(Wsu, WsuT, HD, SED);
  k_tconv<<<dim3(ED / 32, ED / 32, 1), 256, 0, stream>>>(Wsa, WsaT, ED, ED);
  k_tconv<<<dim3(SED / 32, HD / 32, 1), 256, 0, stream>>>(Wsd, WsdT, SED, HD);

  // x -> bf16
  {
    int n = T_TOK * HD;
    k_f2bf<<<(n + 255) / 256, 256, 0, stream>>>(x, xt_bf, n);
  }

  const float* xf_cur = x;  // fp32 activations for the router
  for (int j = 0; j < 2; ++j) {
    // router
    k_router<<<T_TOK / 8, 256, 0, stream>>>(xf_cur, Wg + (long)j * HD * 8, bg + j * 8, wR);
    // zero the combine buffer
    hipMemsetAsync(outF, 0, (size_t)T_TOK * HD * 4, stream);

    // routed experts (dense: weights are exactly 0 for un-routed)
    for (int n = 0; n < NEXP; ++n) {
      k_gemm<EP_STORE><<<dim3(ED / 128, T_TOK / 128), 256, 0, stream>>>(
          xt_bf, HD, WupT + (size_t)n * ED * HD, HD, bup + (long)n * ED,
          (void*)hupB, ED, (const bf16_t*)nullptr, (const float*)nullptr, HD);
      k_gemm<EP_SWIGLU><<<dim3(ED / 128, T_TOK / 128), 256, 0, stream>>>(
          hupB, ED, WswT + (size_t)n * ED * ED, ED, bsw + (long)n * ED,
          (void*)hB, ED, hupB, wR + n, ED);
      k_gemm<EP_ADDOUT><<<dim3(HD / 128, T_TOK / 128), 256, 0, stream>>>(
          hB, ED, WdnT + (size_t)n * HD * ED, ED, bdn + (long)n * HD,
          (void*)outF, HD, (const bf16_t*)nullptr, wR + n, ED);
    }

    // shared experts
    k_gemm<EP_STORE><<<dim3(SED / 128, T_TOK / 128), 256, 0, stream>>>(
        xt_bf, HD, WsuT, HD, bsu, (void*)shB, SED,
        (const bf16_t*)nullptr, (const float*)nullptr, HD);
    // view sh as (2T x E); SwiGLU with shared Wsa
    k_gemm<EP_SWIGLU><<<dim3(ED / 128, (2 * T_TOK) / 128), 256, 0, stream>>>(
        shB, ED, WsaT, ED, bsa, (void*)h2B, ED, shB, (const float*)nullptr, ED);
    // down-project shared (h2 viewed as T x 2048)
    k_gemm<EP_ADDOUT><<<dim3(HD / 128, T_TOK / 128), 256, 0, stream>>>(
        h2B, SED, WsdT, SED, bsd, (void*)outF, HD,
        (const bf16_t*)nullptr, (const float*)nullptr, SED);

    // residual (res is the ORIGINAL x for both chains)
    float* xf_next = (j == 1) ? (float*)d_out : xtF;
    int n = T_TOK * HD;
    k_residual<<<(n + 255) / 256, 256, 0, stream>>>(x, outF, xf_next, xt_bf, n);
    xf_cur = xf_next;
  }
}
